// UNet_24713241821826
// MI455X (gfx1250) — matrix operand order repaired and checked by reading, not compiled, not run on hardware
//
#include <hip/hip_runtime.h>
#include <math.h>

#define NS 16

// ---------------------------------------------------------------------------
// Types for CDNA5 WMMA
// ---------------------------------------------------------------------------
typedef __attribute__((ext_vector_type(16))) __bf16 v16bf;
typedef __attribute__((ext_vector_type(8)))  __bf16 v8bf;
typedef __attribute__((ext_vector_type(8)))  float  v8f;

__device__ __forceinline__ unsigned short f2bf(float f) {
    union { float f; unsigned u; } c; c.f = f;
    unsigned u = c.u;
    unsigned r = (u + 0x7fffu + ((u >> 16) & 1u)) >> 16;   // RNE
    return (unsigned short)r;
}
__device__ __forceinline__ unsigned pk2(float a, float b) {
    return (unsigned)f2bf(a) | ((unsigned)f2bf(b) << 16);
}
__device__ __forceinline__ float siluf(float x) {
    return x / (1.f + __expf(-x));
}

// ---------------------------------------------------------------------------
// GEMM: C[M,N] = A[M,K] (row-major, lda) * B[K,N] (row-major weights)
// bf16 compute via v_wmma_f32_16x16x32_bf16, fp32 accumulate.
// REQUIRES: M % 128 == 0 (true for every call: M = L).
// Block tile 128x128x32, 256 threads = 8 waves (4x2), wave tile 32x64 = 8 WMMA.
// LDS tiles packed as uint (2 x bf16), staged with b128 ops; A fetched with b128, B with 16-bit reads.
// ---------------------------------------------------------------------------
#define BM 128
#define BN 128
#define BK 32

__global__ __launch_bounds__(256) void gemm_bf16_wmma(
    const float* __restrict__ A, int lda, int M, int K,
    const float* __restrict__ B, int N,
    float* __restrict__ C, int ldc,
    const float* __restrict__ bias, int act)
{
    __shared__ __align__(16) unsigned As[BM][BK / 2];   // 128 x 16 uints = 8 KB
    __shared__ __align__(16) unsigned Bs[BK][BN / 2];   // 32 x 64 uints  = 8 KB

    const int tid  = threadIdx.x;
    const int gm0  = blockIdx.y * BM;
    const int gn0  = blockIdx.x * BN;
    const int wave = tid >> 5;
    const int lane = tid & 31;
    const int wm   = wave >> 1;   // 0..3  (row tile)
    const int wn   = wave & 1;    // 0..1  (col tile, 64 wide)

    // staging assignments
    const int arow = tid >> 1;            // 0..127
    const int acb  = (tid & 1) * 16;      // 0 / 16
    const int brow = tid >> 3;            // 0..31
    const int bcb  = (tid & 7) * 16;      // 0..112

    const bool aAligned = ((lda & 3) == 0);
    const bool bAligned = ((N & 3) == 0) && (gn0 + BN <= N);

    v8f acc[2][4];
#pragma unroll
    for (int r = 0; r < 2; ++r)
#pragma unroll
        for (int c = 0; c < 4; ++c)
#pragma unroll
            for (int e = 0; e < 8; ++e) acc[r][c][e] = 0.f;

    for (int k0 = 0; k0 < K; k0 += BK) {
        const bool fullK = (k0 + BK <= K);

        // ---- stage A tile (128 x 32) ----
        {
            float av[16];
            const size_t roff = (size_t)(gm0 + arow) * lda;
            if (fullK && aAligned) {
                const float* ap = A + roff + k0 + acb;
                const float4 p0 = *(const float4*)(ap);
                const float4 p1 = *(const float4*)(ap + 4);
                const float4 p2 = *(const float4*)(ap + 8);
                const float4 p3 = *(const float4*)(ap + 12);
                av[0]=p0.x; av[1]=p0.y; av[2]=p0.z;  av[3]=p0.w;
                av[4]=p1.x; av[5]=p1.y; av[6]=p1.z;  av[7]=p1.w;
                av[8]=p2.x; av[9]=p2.y; av[10]=p2.z; av[11]=p2.w;
                av[12]=p3.x; av[13]=p3.y; av[14]=p3.z; av[15]=p3.w;
                if (k0 + 2 * BK <= K) __builtin_prefetch(ap + BK, 0, 1);
            } else {
#pragma unroll
                for (int i = 0; i < 16; ++i) {
                    int gc = k0 + acb + i;
                    int cc = gc < K ? gc : K - 1;        // clamp: load stays in-bounds
                    float v = A[roff + cc];
                    av[i] = gc < K ? v : 0.f;            // mask, no divergence
                }
            }
            uint4 q0 = make_uint4(pk2(av[0],av[1]),  pk2(av[2],av[3]),
                                  pk2(av[4],av[5]),  pk2(av[6],av[7]));
            uint4 q1 = make_uint4(pk2(av[8],av[9]),  pk2(av[10],av[11]),
                                  pk2(av[12],av[13]), pk2(av[14],av[15]));
            *(uint4*)&As[arow][(acb >> 1)]     = q0;
            *(uint4*)&As[arow][(acb >> 1) + 4] = q1;
        }
        // ---- stage B tile (32 x 128) ----
        {
            float bv[16];
            const int gr = k0 + brow;
            if (fullK && bAligned) {
                const float* bp = B + (size_t)gr * N + gn0 + bcb;
                const float4 p0 = *(const float4*)(bp);
                const float4 p1 = *(const float4*)(bp + 4);
                const float4 p2 = *(const float4*)(bp + 8);
                const float4 p3 = *(const float4*)(bp + 12);
                bv[0]=p0.x; bv[1]=p0.y; bv[2]=p0.z;  bv[3]=p0.w;
                bv[4]=p1.x; bv[5]=p1.y; bv[6]=p1.z;  bv[7]=p1.w;
                bv[8]=p2.x; bv[9]=p2.y; bv[10]=p2.z; bv[11]=p2.w;
                bv[12]=p3.x; bv[13]=p3.y; bv[14]=p3.z; bv[15]=p3.w;
            } else {
                const int rr = gr < K ? gr : K - 1;
                const size_t roff = (size_t)rr * N;
#pragma unroll
                for (int i = 0; i < 16; ++i) {
                    int gc = gn0 + bcb + i;
                    int cc = gc < N ? gc : N - 1;
                    float v = B[roff + cc];
                    bv[i] = (gr < K && gc < N) ? v : 0.f;
                }
            }
            uint4 q0 = make_uint4(pk2(bv[0],bv[1]),  pk2(bv[2],bv[3]),
                                  pk2(bv[4],bv[5]),  pk2(bv[6],bv[7]));
            uint4 q1 = make_uint4(pk2(bv[8],bv[9]),  pk2(bv[10],bv[11]),
                                  pk2(bv[12],bv[13]), pk2(bv[14],bv[15]));
            *(uint4*)&Bs[brow][(bcb >> 1)]     = q0;
            *(uint4*)&Bs[brow][(bcb >> 1) + 4] = q1;
        }
        __syncthreads();

        // ---- WMMA fragments (ISA 7.12.2 layouts, wave32) ----
        union U { uint4 q; v8bf h; };
        const int kb2 = (lane >> 4) * 4;          // packed-uint K base (0 / 4)
        const int mrb = wm * 32 + (lane & 15);    // A: lane -> row
        v16bf afrag[2], bfrag[4];
#pragma unroll
        for (int r = 0; r < 2; ++r) {
            U u0, u1;
            u0.q = *(const uint4*)&As[mrb + r * 16][kb2];
            u1.q = *(const uint4*)&As[mrb + r * 16][8 + kb2];
#pragma unroll
            for (int i = 0; i < 8; ++i) { afrag[r][i] = u0.h[i]; afrag[r][i + 8] = u1.h[i]; }
        }
#pragma unroll
        for (int c = 0; c < 4; ++c) {
            const int ncol = wn * 64 + c * 16 + (lane & 15);   // B: lane -> column
            const unsigned short* bsh = (const unsigned short*)&Bs[0][0];
#pragma unroll
            for (int i = 0; i < 8; ++i) {
                bfrag[c][i]     = __builtin_bit_cast(__bf16, bsh[(2 * kb2 + i) * BN + ncol]);
                bfrag[c][i + 8] = __builtin_bit_cast(__bf16, bsh[(16 + 2 * kb2 + i) * BN + ncol]);
            }
        }
#pragma unroll
        for (int r = 0; r < 2; ++r)
#pragma unroll
            for (int c = 0; c < 4; ++c)
                acc[r][c] = __builtin_amdgcn_wmma_f32_16x16x32_bf16(
                    false, afrag[r], false, bfrag[c], (short)0, acc[r][c], false, false);
        __syncthreads();
    }

    // ---- store C (+bias, +activation). C/D layout: VGPR v -> M, lane -> N ----
    const int nloc  = lane & 15;
    const int mhalf = (lane >> 4) * 8;
#pragma unroll
    for (int r = 0; r < 2; ++r)
#pragma unroll
        for (int c = 0; c < 4; ++c) {
            int col = gn0 + wn * 64 + c * 16 + nloc;
            if (col < N) {
                float bval = bias ? bias[col] : 0.f;
#pragma unroll
                for (int v = 0; v < 8; ++v) {
                    int row = gm0 + wm * 32 + r * 16 + mhalf + v;   // row < M guaranteed
                    float val = acc[r][c][v] + bval;
                    if (act == 1) val = (val > 20.f) ? val : log1pf(__expf(val));
                    C[(size_t)row * ldc + col] = val;
                }
            }
        }
}

// ---------------------------------------------------------------------------
// Elementwise / structural kernels
// ---------------------------------------------------------------------------
__global__ void patchify_k(const float* __restrict__ inp, float* __restrict__ xs) {
    int idx = blockIdx.x * 256 + threadIdx.x;
    if (idx >= 65536) return;
    int n = idx >> 10, h = (idx >> 5) & 31, w = idx & 31;
    int i8 = n >> 3, j8 = n & 7;
    xs[(size_t)idx * 4] = inp[(i8 * 32 + h) * 256 + (j8 * 32 + w)];   // ld = 4 (padded)
}

__global__ void unpatchify_k(const float* __restrict__ y, float* __restrict__ out) {
    int idx = blockIdx.x * 256 + threadIdx.x;
    if (idx >= 4 * 65536) return;
    int c = idx >> 16, r = idx & 65535;
    int n = r >> 10, h = (r >> 5) & 31, w = r & 31;
    int i8 = n >> 3, j8 = n & 7;
    out[(size_t)c * 65536 + (size_t)(i8 * 32 + h) * 256 + (j8 * 32 + w)] = y[(size_t)r * 4 + c];
}

// depthwise causal conv1d (k=4, pad 3 left) + SiLU.  xz strided (ld), xi strided (ldo).
__global__ void conv_silu_k(const float* __restrict__ xz, int ld,
                            const float* __restrict__ w, const float* __restrict__ b,
                            float* __restrict__ xi, int ldo, int L, int d_in) {
    int idx = blockIdx.x * 256 + threadIdx.x;
    if (idx >= L * d_in) return;
    int c = idx % d_in, l = idx / d_in;
    float acc = b[c];
#pragma unroll
    for (int k = 0; k < 4; ++k) {
        int t = l - 3 + k;
        if (t >= 0) acc += w[c * 4 + k] * xz[(size_t)t * ld + c];
    }
    xi[(size_t)l * ldo + c] = siluf(acc);
}

// Selective scan: thread = (channel d, state n); 16-lane shfl_xor reduction.
__global__ void scan_k(const float* __restrict__ xi, int ldu,
                       const float* __restrict__ delta, int ldd,
                       const float* __restrict__ xdbl, int ldx, int dtr,
                       const float* __restrict__ A_log, const float* __restrict__ Dp,
                       float* __restrict__ y, int ldy, int L, int d_in) {
    int gid = blockIdx.x * 256 + threadIdx.x;
    int d = gid >> 4, n = gid & 15;
    bool active = d < d_in;
    float A  = active ? -__expf(A_log[(size_t)d * NS + n]) : 0.f;
    float Dv = active ? Dp[d] : 0.f;
    float h = 0.f;
    for (int l = 0; l < L; ++l) {
        float dt = active ? delta[(size_t)l * ldd + d] : 0.f;
        float u  = active ? xi[(size_t)l * ldu + d] : 0.f;
        float Bt = active ? xdbl[(size_t)l * ldx + dtr + n] : 0.f;
        float Ct = active ? xdbl[(size_t)l * ldx + dtr + NS + n] : 0.f;
        h = __expf(dt * A) * h + dt * u * Bt;
        float p = h * Ct;
        p += __shfl_xor(p, 1, 16);
        p += __shfl_xor(p, 2, 16);
        p += __shfl_xor(p, 4, 16);
        p += __shfl_xor(p, 8, 16);
        if (active && n == 0) y[(size_t)l * ldy + d] = p + u * Dv;
    }
}

// y *= silu(z), z = xz[:, d_in:]
__global__ void gate_k(float* __restrict__ y, int ldy, const float* __restrict__ xz,
                       int ldxz, int d_in, int L) {
    int idx = blockIdx.x * 256 + threadIdx.x;
    if (idx >= L * d_in) return;
    int d = idx % d_in, l = idx / d_in;
    float z = xz[(size_t)l * ldxz + d_in + d];
    y[(size_t)l * ldy + d] *= siluf(z);
}

__global__ __launch_bounds__(256) void ln_k(float* __restrict__ x,
                                            const float* __restrict__ g,
                                            const float* __restrict__ b, int C) {
    __shared__ float ssum[256], ssq[256];
    int tid = threadIdx.x;
    size_t base = (size_t)blockIdx.x * C;
    float s = 0.f, sq = 0.f;
    for (int c = tid; c < C; c += 256) { float v = x[base + c]; s += v; sq += v * v; }
    ssum[tid] = s; ssq[tid] = sq;
    __syncthreads();
    for (int off = 128; off > 0; off >>= 1) {
        if (tid < off) { ssum[tid] += ssum[tid + off]; ssq[tid] += ssq[tid + off]; }
        __syncthreads();
    }
    float mean = ssum[0] / C;
    float var  = ssq[0] / C - mean * mean;
    float rstd = rsqrtf(var + 1e-5f);
    for (int c = tid; c < C; c += 256)
        x[base + c] = (x[base + c] - mean) * rstd * g[c] + b[c];
}

__global__ void add_k(float* __restrict__ o, const float* __restrict__ a,
                      const float* __restrict__ b, int n) {
    int i = blockIdx.x * 256 + threadIdx.x;
    if (i < n) o[i] = a[i] + b[i];
}

__global__ void avgpool_k(const float* __restrict__ x, float* __restrict__ o,
                          int C, int nblk, int h, int w) {
    int ho = h >> 1, wo = w >> 1;
    int total = nblk * ho * wo * C;
    int idx = blockIdx.x * 256 + threadIdx.x;
    if (idx >= total) return;
    int c = idx % C, l = idx / C;
    int j = l % wo, t = l / wo, i = t % ho, n = t / ho;
    size_t base = ((size_t)(n * h + 2 * i) * w + 2 * j) * C + c;
    size_t rs = (size_t)w * C;
    o[(size_t)l * C + c] = 0.25f * (x[base] + x[base + C] + x[base + rs] + x[base + rs + C]);
}

__global__ void upsample_k(const float* __restrict__ x, float* __restrict__ o,
                           int C, int nblk, int h, int w) {
    int ho = 2 * h, wo = 2 * w;
    int total = nblk * ho * wo * C;
    int idx = blockIdx.x * 256 + threadIdx.x;
    if (idx >= total) return;
    int c = idx % C, l = idx / C;
    int j = l % wo, t = l / wo, i = t % ho, n = t / ho;
    float fi = (h > 1) ? i * ((float)(h - 1) / (float)(ho - 1)) : 0.f;
    float fj = (w > 1) ? j * ((float)(w - 1) / (float)(wo - 1)) : 0.f;
    int il = (int)floorf(fi), ih = min(il + 1, h - 1);
    int jl = (int)floorf(fj), jh = min(jl + 1, w - 1);
    float wi = fi - il, wj = fj - jl;
#define XAT(r, cc) x[((size_t)(n * h + (r)) * w + (cc)) * C + c]
    float v = (1.f - wi) * ((1.f - wj) * XAT(il, jl) + wj * XAT(il, jh)) +
              wi * ((1.f - wj) * XAT(ih, jl) + wj * XAT(ih, jh));
#undef XAT
    o[(size_t)l * C + c] = v;
}

__global__ void concat_k(const float* __restrict__ a, int Ca,
                         const float* __restrict__ b, int Cb,
                         float* __restrict__ o, int L) {
    int C = Ca + Cb;
    int idx = blockIdx.x * 256 + threadIdx.x;
    if (idx >= L * C) return;
    int c = idx % C, l = idx / C;
    o[idx] = (c < Ca) ? a[(size_t)l * Ca + c] : b[(size_t)l * Cb + (c - Ca)];
}

// ---------------------------------------------------------------------------
// Host-side orchestration
// ---------------------------------------------------------------------------
namespace {

struct Alloc {
    char*  base;
    size_t off;
    float* take(size_t nfloat) {
        float* p = (float*)(base + off);
        off += ((nfloat * sizeof(float) + 255) / 256) * 256;
        return p;
    }
};

struct SSM {
    const float *A_log, *D, *W_dt, *W_in, *W_out, *W_x, *b_dt, *conv_b, *conv_w;
    int d_model, d_in, dtr, out_c;
};
struct DC {
    const float *b1, *b2, *g1, *g2, *res_W, *res_b, *res_be, *res_g;
    SSM s1, s2;
    int cin, cout;
};

SSM parse_ssm(void* const* p, int& i, int d_model, int out_c) {
    SSM s;
    s.A_log  = (const float*)p[i++];
    s.D      = (const float*)p[i++];
    s.W_dt   = (const float*)p[i++];
    s.W_in   = (const float*)p[i++];
    s.W_out  = (const float*)p[i++];
    s.W_x    = (const float*)p[i++];
    s.b_dt   = (const float*)p[i++];
    s.conv_b = (const float*)p[i++];
    s.conv_w = (const float*)p[i++];
    s.d_model = d_model;
    s.d_in    = 2 * d_model;
    s.dtr     = (d_model + 15) / 16; if (s.dtr < 1) s.dtr = 1;
    s.out_c   = out_c;
    return s;
}

DC parse_dc(void* const* p, int& i, int cin, int cout) {
    DC d;
    d.b1     = (const float*)p[i++];
    d.b2     = (const float*)p[i++];
    d.g1     = (const float*)p[i++];
    d.g2     = (const float*)p[i++];
    d.res_W  = (const float*)p[i++];
    d.res_b  = (const float*)p[i++];
    d.res_be = (const float*)p[i++];
    d.res_g  = (const float*)p[i++];
    d.s1 = parse_ssm(p, i, cin, cout);
    d.s2 = parse_ssm(p, i, cout, cout);
    d.cin = cin; d.cout = cout;
    return d;
}

inline int ld4(int n) { return (n + 3) & ~3; }
inline dim3 eg(long long total) { return dim3((unsigned)((total + 255) / 256)); }

void gemm(const float* A, int lda, int M, int K, const float* B, int N,
          float* C, int ldc, const float* bias, int act, hipStream_t st) {
    dim3 grid((N + BN - 1) / BN, (M + BM - 1) / BM);
    gemm_bf16_wmma<<<grid, 256, 0, st>>>(A, lda, M, K, B, N, C, ldc, bias, act);
}

void run_mamba(const SSM& s, const float* x, int ldx, int L, float* out, int ldo,
               Alloc ws, hipStream_t st) {
    const int d_in = s.d_in, dtr = s.dtr;
    const int nx = dtr + 2 * NS;
    const int ldxz = 2 * d_in;                       // >= 4, multiple of 4
    float* xz = ws.take((size_t)L * ldxz);
    gemm(x, ldx, L, s.d_model, s.W_in, 2 * d_in, xz, ldxz, nullptr, 0, st);

    const int ldxi = ld4(d_in);
    float* xi = ws.take((size_t)L * ldxi);
    conv_silu_k<<<eg((long long)L * d_in), 256, 0, st>>>(
        xz, ldxz, s.conv_w, s.conv_b, xi, ldxi, L, d_in);

    const int ldxd = ld4(nx);
    float* xdbl = ws.take((size_t)L * ldxd);
    gemm(xi, ldxi, L, d_in, s.W_x, nx, xdbl, ldxd, nullptr, 0, st);

    const int lddl = ld4(d_in);
    float* delta = ws.take((size_t)L * lddl);
    gemm(xdbl, ldxd, L, dtr, s.W_dt, d_in, delta, lddl, s.b_dt, /*softplus*/ 1, st);

    const int ldy = ld4(d_in);
    float* y = ws.take((size_t)L * ldy);
    scan_k<<<eg((long long)d_in * 16), 256, 0, st>>>(
        xi, ldxi, delta, lddl, xdbl, ldxd, dtr, s.A_log, s.D, y, ldy, L, d_in);
    gate_k<<<eg((long long)L * d_in), 256, 0, st>>>(y, ldy, xz, ldxz, d_in, L);

    gemm(y, ldy, L, d_in, s.W_out, s.out_c, out, ldo, nullptr, 0, st);
}

void run_dc(const DC& p, const float* x, int ldx, int L, float* out, Alloc ws,
            hipStream_t st) {
    const int cout = p.cout;                         // always multiple of 4
    float* res = ws.take((size_t)L * cout);
    gemm(x, ldx, L, p.cin, p.res_W, cout, res, cout, p.res_b, 0, st);
    ln_k<<<L, 256, 0, st>>>(res, p.res_g, p.res_be, cout);
    float* h1 = ws.take((size_t)L * cout);
    run_mamba(p.s1, x, ldx, L, h1, cout, ws, st);
    ln_k<<<L, 256, 0, st>>>(h1, p.g1, p.b1, cout);
    float* h2 = ws.take((size_t)L * cout);
    run_mamba(p.s2, h1, cout, L, h2, cout, ws, st);
    ln_k<<<L, 256, 0, st>>>(h2, p.g2, p.b2, cout);
    add_k<<<eg((long long)L * cout), 256, 0, st>>>(out, res, h2, L * cout);
}

} // namespace

extern "C" void kernel_launch(void* const* d_in, const int* in_sizes, int n_in,
                              void* d_out, int out_size, void* d_ws, size_t ws_size,
                              hipStream_t stream) {
    (void)in_sizes; (void)n_in; (void)out_size; (void)ws_size;
    const float* inp = (const float*)d_in[0];

    // Pytree (sorted-key) parameter order: down1..down5, embd, out, up1..up5
    int i = 1;
    DC down1 = parse_dc(d_in, i,   64,  128);
    DC down2 = parse_dc(d_in, i,  128,  256);
    DC down3 = parse_dc(d_in, i,  256,  512);
    DC down4 = parse_dc(d_in, i,  512, 1024);
    DC down5 = parse_dc(d_in, i, 1024, 2048);
    DC embd  = parse_dc(d_in, i,    1,   64);
    DC outp  = parse_dc(d_in, i,   64,    4);
    DC up1   = parse_dc(d_in, i, 3072, 1024);
    DC up2   = parse_dc(d_in, i, 1536,  512);
    DC up3   = parse_dc(d_in, i,  768,  256);
    DC up4   = parse_dc(d_in, i,  384,  128);
    DC up5   = parse_dc(d_in, i,  192,   64);

    Alloc ws{(char*)d_ws, 0};
    hipStream_t st = stream;

    // input patchify: (1,1,256,256) -> seq (65536, ld=4, 1 valid col)
    float* xs = ws.take((size_t)65536 * 4);
    patchify_k<<<eg(65536), 256, 0, st>>>(inp, xs);

    // encoder
    float* x0 = ws.take((size_t)65536 * 64);
    run_dc(embd, xs, 4, 65536, x0, ws, st);

    float* t1 = ws.take((size_t)65536 * 128);
    run_dc(down1, x0, 64, 65536, t1, ws, st);
    float* x1 = ws.take((size_t)16384 * 128);
    avgpool_k<<<eg((long long)16384 * 128), 256, 0, st>>>(t1, x1, 128, 64, 32, 32);

    float* t2 = ws.take((size_t)16384 * 256);
    run_dc(down2, x1, 128, 16384, t2, ws, st);
    float* x2 = ws.take((size_t)4096 * 256);
    avgpool_k<<<eg((long long)4096 * 256), 256, 0, st>>>(t2, x2, 256, 64, 16, 16);

    float* t3 = ws.take((size_t)4096 * 512);
    run_dc(down3, x2, 256, 4096, t3, ws, st);
    float* x3 = ws.take((size_t)1024 * 512);
    avgpool_k<<<eg((long long)1024 * 512), 256, 0, st>>>(t3, x3, 512, 64, 8, 8);

    float* t4 = ws.take((size_t)1024 * 1024);
    run_dc(down4, x3, 512, 1024, t4, ws, st);
    float* x4 = ws.take((size_t)256 * 1024);
    avgpool_k<<<eg((long long)256 * 1024), 256, 0, st>>>(t4, x4, 1024, 64, 4, 4);

    float* t5 = ws.take((size_t)256 * 2048);
    run_dc(down5, x4, 1024, 256, t5, ws, st);
    float* x5 = ws.take((size_t)64 * 2048);
    avgpool_k<<<eg((long long)64 * 2048), 256, 0, st>>>(t5, x5, 2048, 64, 2, 2);

    // decoder
    float* u1 = ws.take((size_t)256 * 2048);
    upsample_k<<<eg((long long)256 * 2048), 256, 0, st>>>(x5, u1, 2048, 64, 1, 1);
    float* c1 = ws.take((size_t)256 * 3072);
    concat_k<<<eg((long long)256 * 3072), 256, 0, st>>>(u1, 2048, x4, 1024, c1, 256);
    float* y1 = ws.take((size_t)256 * 1024);
    run_dc(up1, c1, 3072, 256, y1, ws, st);

    float* u2 = ws.take((size_t)1024 * 1024);
    upsample_k<<<eg((long long)1024 * 1024), 256, 0, st>>>(y1, u2, 1024, 64, 2, 2);
    float* c2 = ws.take((size_t)1024 * 1536);
    concat_k<<<eg((long long)1024 * 1536), 256, 0, st>>>(u2, 1024, x3, 512, c2, 1024);
    float* y2 = ws.take((size_t)1024 * 512);
    run_dc(up2, c2, 1536, 1024, y2, ws, st);

    float* u3 = ws.take((size_t)4096 * 512);
    upsample_k<<<eg((long long)4096 * 512), 256, 0, st>>>(y2, u3, 512, 64, 4, 4);
    float* c3 = ws.take((size_t)4096 * 768);
    concat_k<<<eg((long long)4096 * 768), 256, 0, st>>>(u3, 512, x2, 256, c3, 4096);
    float* y3 = ws.take((size_t)4096 * 256);
    run_dc(up3, c3, 768, 4096, y3, ws, st);

    float* u4 = ws.take((size_t)16384 * 256);
    upsample_k<<<eg((long long)16384 * 256), 256, 0, st>>>(y3, u4, 256, 64, 8, 8);
    float* c4 = ws.take((size_t)16384 * 384);
    concat_k<<<eg((long long)16384 * 384), 256, 0, st>>>(u4, 256, x1, 128, c4, 16384);
    float* y4 = ws.take((size_t)16384 * 128);
    run_dc(up4, c4, 384, 16384, y4, ws, st);

    float* u5 = ws.take((size_t)65536 * 128);
    upsample_k<<<eg((long long)65536 * 128), 256, 0, st>>>(y4, u5, 128, 64, 16, 16);
    float* c5 = ws.take((size_t)65536 * 192);
    concat_k<<<eg((long long)65536 * 192), 256, 0, st>>>(u5, 128, x0, 64, c5, 65536);
    float* y5 = ws.take((size_t)65536 * 64);
    run_dc(up5, c5, 192, 65536, y5, ws, st);

    float* yo = ws.take((size_t)65536 * 4);
    run_dc(outp, y5, 64, 65536, yo, ws, st);

    unpatchify_k<<<eg((long long)4 * 65536), 256, 0, st>>>(yo, (float*)d_out);
}
